// GatFCM_42021960024201
// MI455X (gfx1250) — compile-verified, matrix operand order checked by reading, not run on hardware
//
#include <hip/hip_runtime.h>
#include <hip/hip_bf16.h>

#define N_NODES 50000
#define N_EDGES 800000
#define D_IN    1028
#define D_OUT   256
#define K_PAD   1056           // D_IN padded up to multiple of 32
#define NW      512            // fused N dimension: [Wl | Wr]
#define K_TOT   (N_EDGES + N_NODES)  // edges + self loops = 850000

typedef __attribute__((ext_vector_type(16))) __bf16 v16bf;
typedef __attribute__((ext_vector_type(8)))  __bf16 v8bf;
typedef __attribute__((ext_vector_type(8)))  float  v8f;

// ---------- order-preserving float <-> uint key for atomicMax ----------
__device__ __forceinline__ unsigned fkey(float f) {
    unsigned u = __float_as_uint(f);
    return (u & 0x80000000u) ? ~u : (u | 0x80000000u);
}
__device__ __forceinline__ float funkey(unsigned k) {
    return __uint_as_float((k & 0x80000000u) ? (k & 0x7FFFFFFFu) : ~k);
}

__device__ __forceinline__ v16bf cat16(v8bf b0, v8bf b1) {
    v16bf b;
#pragma unroll
    for (int i = 0; i < 8; ++i) { b[i] = b0[i]; b[8 + i] = b1[i]; }
    return b;
}

__device__ __forceinline__ v16bf cvt_a(float4 a0, float4 a1, float4 a2, float4 a3) {
    v16bf a;
    a[0]=(__bf16)a0.x; a[1]=(__bf16)a0.y; a[2]=(__bf16)a0.z; a[3]=(__bf16)a0.w;
    a[4]=(__bf16)a1.x; a[5]=(__bf16)a1.y; a[6]=(__bf16)a1.z; a[7]=(__bf16)a1.w;
    a[8]=(__bf16)a2.x; a[9]=(__bf16)a2.y; a[10]=(__bf16)a2.z; a[11]=(__bf16)a2.w;
    a[12]=(__bf16)a3.x; a[13]=(__bf16)a3.y; a[14]=(__bf16)a3.z; a[15]=(__bf16)a3.w;
    return a;
}

// ---------- prep: Wt[n][k] = transpose([Wl|Wr]) in bf16, zero-padded K ----------
__global__ void k_prep_w(const float* __restrict__ Wl, const float* __restrict__ Wr,
                         __bf16* __restrict__ Wt) {
    int idx = blockIdx.x * blockDim.x + threadIdx.x;
    if (idx >= NW * K_PAD) return;
    int n = idx / K_PAD;
    int k = idx - n * K_PAD;
    float v = 0.0f;
    if (k < D_IN)
        v = (n < D_OUT) ? Wl[(size_t)k * D_OUT + n] : Wr[(size_t)k * D_OUT + (n - D_OUT)];
    Wt[idx] = (__bf16)v;
}

__global__ void k_init(unsigned* __restrict__ emax_u, float* __restrict__ denom) {
    int i = blockIdx.x * blockDim.x + threadIdx.x;
    if (i < N_NODES) { emax_u[i] = 0u; denom[i] = 0.0f; }  // key 0 decodes below all reals
}

__global__ void k_out_bias(const float* __restrict__ bias, float* __restrict__ out) {
    int i = blockIdx.x * blockDim.x + threadIdx.x;
    if (i < N_NODES * D_OUT) out[i] = bias[i & (D_OUT - 1)];
}

// ---------- fused GEMM: [xl | xr] = x @ [Wl | Wr], bf16 WMMA, f32 acc ----------
// block = 8 waves; wave w -> rows [blockIdx.x*128 + w*16, +16), cols blockIdx.y*128..+128
// software-pipelined one k-step deep: B fragments for step ks+1 are issued right
// after each b[t] is consumed by its WMMA; A raw loads issued before the burst.
__global__ __launch_bounds__(256)
void k_gemm(const float* __restrict__ x, const __bf16* __restrict__ Wt,
            float* __restrict__ xl, float* __restrict__ xr) {
    const int lane = threadIdx.x & 31;
    const int wave = threadIdx.x >> 5;
    const int half = lane >> 4;       // 0: K {0-7,16-23}; 1: K {8-15,24-31}
    const int r16  = lane & 15;
    const int nb   = blockIdx.y;      // 0..3 (cols 0..511 in tiles of 128)
    const int rowbase = blockIdx.x * 128 + wave * 16;

    const int row  = rowbase + r16;
    const int rowc = row < N_NODES ? row : (N_NODES - 1);   // clamp reads; stores guarded
    const float*  xrow  = x  + (size_t)rowc * D_IN;
    const __bf16* wbase = Wt + (size_t)(nb * 128 + r16) * K_PAD;
    const int kg_off = half << 3;

    v8f zero = {};
    v8f acc[8];
#pragma unroll
    for (int t = 0; t < 8; ++t) acc[t] = zero;

    // ---- prologue: A + all 8 B fragments for k-step 0 ----
    v16bf b[8];
#pragma unroll
    for (int t = 0; t < 8; ++t) {
        const __bf16* wp = wbase + (size_t)(t * 16) * K_PAD + kg_off;
        b[t] = cat16(*(const v8bf*)(wp), *(const v8bf*)(wp + 16));
    }
    v16bf a = cvt_a(*(const float4*)(xrow + kg_off),
                    *(const float4*)(xrow + kg_off + 4),
                    *(const float4*)(xrow + kg_off + 16),
                    *(const float4*)(xrow + kg_off + 20));

    // ---- steady state: ks = 0..30, prefetching ks+1 ----
#pragma unroll 1
    for (int ks = 0; ks < 31; ++ks) {
        const int kgn = ks * 32 + 32 + kg_off;      // next k-step's group base
        const float4 n0 = *(const float4*)(xrow + kgn);
        const float4 n1 = *(const float4*)(xrow + kgn + 4);
        const float4 n2 = *(const float4*)(xrow + kgn + 16);
        const float4 n3 = *(const float4*)(xrow + kgn + 20);
#pragma unroll
        for (int t = 0; t < 8; ++t) {
            acc[t] = __builtin_amdgcn_wmma_f32_16x16x32_bf16(
                         false, a, false, b[t], (short)0, acc[t], false, false);
            const __bf16* wp = wbase + (size_t)(t * 16) * K_PAD + kgn;
            b[t] = cat16(*(const v8bf*)(wp), *(const v8bf*)(wp + 16));
        }
        a = cvt_a(n0, n1, n2, n3);
    }

    // ---- ks = 31: consume, preload B for the tail step (Wt zero-padded) ----
    {
        const int kgn = 1024 + kg_off;
#pragma unroll
        for (int t = 0; t < 8; ++t) {
            acc[t] = __builtin_amdgcn_wmma_f32_16x16x32_bf16(
                         false, a, false, b[t], (short)0, acc[t], false, false);
            const __bf16* wp = wbase + (size_t)(t * 16) * K_PAD + kgn;
            b[t] = cat16(*(const v8bf*)(wp), *(const v8bf*)(wp + 16));
        }
    }

    // ---- tail step: K 1024..1055, only 1024..1027 valid in x ----
    {
        const int kg0 = 1024 + kg_off;
        v16bf at;
#pragma unroll
        for (int i = 0; i < 16; ++i) {
            int kk = kg0 + ((i < 8) ? i : (i - 8) + 16);
            at[i] = (kk < D_IN) ? (__bf16)xrow[kk] : (__bf16)0.0f;
        }
#pragma unroll
        for (int t = 0; t < 8; ++t)
            acc[t] = __builtin_amdgcn_wmma_f32_16x16x32_bf16(
                         false, at, false, b[t], (short)0, acc[t], false, false);
    }

    // store: C/D layout -> lane<16: rows v, N=lane; lane>=16: rows v+8, N=lane-16
    float* dst = (nb < 2) ? xl : xr;
    const int colbase = (nb & 1) * 128;
#pragma unroll
    for (int t = 0; t < 8; ++t) {
        const int col = colbase + t * 16 + r16;
#pragma unroll
        for (int v = 0; v < 8; ++v) {
            const int r = rowbase + v + half * 8;
            if (r < N_NODES) dst[(size_t)r * D_OUT + col] = acc[t][v];
        }
    }
}

// ---------- edge scoring: one wave32 per edge ----------
__global__ void k_edge_e(const long long* __restrict__ ei,
                         const float* __restrict__ xl, const float* __restrict__ xr,
                         const float* __restrict__ att,
                         float* __restrict__ e, unsigned* __restrict__ emax_u) {
    int gid = blockIdx.x * blockDim.x + threadIdx.x;
    int k = gid >> 5, lane = gid & 31;
    if (k >= K_TOT) return;
    int s, d;
    if (k < N_EDGES) { s = (int)ei[k]; d = (int)ei[N_EDGES + k]; }
    else             { s = k - N_EDGES; d = s; }
    const float4* pl = (const float4*)(xl + (size_t)s * D_OUT) + lane * 2;
    const float4* pr = (const float4*)(xr + (size_t)d * D_OUT) + lane * 2;
    const float4* pa = (const float4*)att + lane * 2;
    float sum = 0.0f;
#pragma unroll
    for (int g = 0; g < 2; ++g) {
        float4 ml = pl[g], mr = pr[g], aa = pa[g];
        float m0 = ml.x + mr.x, m1 = ml.y + mr.y, m2 = ml.z + mr.z, m3 = ml.w + mr.w;
        m0 = m0 > 0.f ? m0 : 0.2f * m0;  m1 = m1 > 0.f ? m1 : 0.2f * m1;
        m2 = m2 > 0.f ? m2 : 0.2f * m2;  m3 = m3 > 0.f ? m3 : 0.2f * m3;
        sum += m0 * aa.x + m1 * aa.y + m2 * aa.z + m3 * aa.w;
    }
#pragma unroll
    for (int off = 16; off > 0; off >>= 1) sum += __shfl_xor(sum, off, 32);
    if (lane == 0) {
        e[k] = sum;
        atomicMax(emax_u + d, fkey(sum));
    }
}

// ---------- softmax numerator + denominator ----------
__global__ void k_edge_ex(const long long* __restrict__ ei,
                          const unsigned* __restrict__ emax_u,
                          float* __restrict__ e, float* __restrict__ denom) {
    int k = blockIdx.x * blockDim.x + threadIdx.x;
    if (k >= K_TOT) return;
    int d = (k < N_EDGES) ? (int)ei[N_EDGES + k] : (k - N_EDGES);
    float ex = __expf(e[k] - funkey(emax_u[d]));
    e[k] = ex;                       // in-place: e now holds exp(e - emax[dst])
    atomicAdd(denom + d, ex);
}

// ---------- weighted scatter: out[dst] += alpha * xl[src] ----------
__global__ void k_edge_scatter(const long long* __restrict__ ei,
                               const float* __restrict__ xl,
                               const float* __restrict__ e,
                               const float* __restrict__ denom,
                               float* __restrict__ out) {
    int gid = blockIdx.x * blockDim.x + threadIdx.x;
    int k = gid >> 5, lane = gid & 31;
    if (k >= K_TOT) return;
    int s, d;
    if (k < N_EDGES) { s = (int)ei[k]; d = (int)ei[N_EDGES + k]; }
    else             { s = k - N_EDGES; d = s; }
    float alpha = e[k] / denom[d];
    const float4* pl = (const float4*)(xl + (size_t)s * D_OUT) + lane * 2;
    float* po = out + (size_t)d * D_OUT + lane * 8;
#pragma unroll
    for (int g = 0; g < 2; ++g) {
        float4 v = pl[g];
        atomicAdd(po + g * 4 + 0, alpha * v.x);
        atomicAdd(po + g * 4 + 1, alpha * v.y);
        atomicAdd(po + g * 4 + 2, alpha * v.z);
        atomicAdd(po + g * 4 + 3, alpha * v.w);
    }
}

extern "C" void kernel_launch(void* const* d_in, const int* in_sizes, int n_in,
                              void* d_out, int out_size, void* d_ws, size_t ws_size,
                              hipStream_t stream) {
    const float*     x    = (const float*)d_in[0];
    const long long* ei   = (const long long*)d_in[1];   // int64 edge_index [2, E]
    const float*     Wl   = (const float*)d_in[2];
    const float*     Wr   = (const float*)d_in[3];
    const float*     att  = (const float*)d_in[4];
    const float*     bias = (const float*)d_in[5];
    float*           out  = (float*)d_out;

    char* ws = (char*)d_ws;
    size_t off = 0;
    auto take = [&](size_t bytes) {
        char* p = ws + off;
        off += (bytes + 255) & ~(size_t)255;
        return p;
    };
    float*    xl     = (float*)   take((size_t)N_NODES * D_OUT * sizeof(float));
    float*    xr     = (float*)   take((size_t)N_NODES * D_OUT * sizeof(float));
    __bf16*   Wt     = (__bf16*)  take((size_t)NW * K_PAD * sizeof(__bf16));
    float*    e      = (float*)   take((size_t)K_TOT * sizeof(float));
    unsigned* emax_u = (unsigned*)take((size_t)N_NODES * sizeof(unsigned));
    float*    denom  = (float*)   take((size_t)N_NODES * sizeof(float));

    k_prep_w      <<<(NW * K_PAD + 255) / 256, 256, 0, stream>>>(Wl, Wr, Wt);
    k_init        <<<(N_NODES + 255) / 256,    256, 0, stream>>>(emax_u, denom);
    k_out_bias    <<<(N_NODES * D_OUT + 255) / 256, 256, 0, stream>>>(bias, out);
    k_gemm        <<<dim3((N_NODES + 127) / 128, 4), 256, 0, stream>>>(x, Wt, xl, xr);
    k_edge_e      <<<(K_TOT * 32 + 255) / 256, 256, 0, stream>>>(ei, xl, xr, att, e, emax_u);
    k_edge_ex     <<<(K_TOT + 255) / 256,      256, 0, stream>>>(ei, emax_u, e, denom);
    k_edge_scatter<<<(K_TOT * 32 + 255) / 256, 256, 0, stream>>>(ei, xl, e, denom, out);
}
